// MoELayer_75445395521789
// MI455X (gfx1250) — compile-verified, matrix operand order checked by reading, not run on hardware
//
#include <hip/hip_runtime.h>
#include <hip/hip_bf16.h>
#include <math.h>

// ---------------- problem constants ----------------
#define H_DIM  1024
#define I_DIM  2048
#define E_NUM  8
#define T_TOK  8192        // B*S = 4*2048

// ---------------- WMMA types (CDNA5 / gfx1250, wave32) ----------------
typedef __attribute__((ext_vector_type(16))) __bf16 v16bf;
typedef __attribute__((ext_vector_type(8)))  float  v8f;

union FragU { v16bf v; uint4 q[2]; };

// Load a 16x32 bf16 fragment (A-layout) or 32x16 (B-layout, stored transposed
// [n][k]) from an LDS region laid out [row][32] bf16, row stride 64B.
// Per CDNA5 ISA 7.12.2: lanes 0-15 hold K {0..7 | 16..23}, lanes 16-31 hold
// K {8..15 | 24..31}, row/col = lane&15.
__device__ __forceinline__ v16bf ld_frag16(const __bf16* base, int tile) {
  const int lane = threadIdx.x & 31;
  const int kk = (lane & 16) ? 8 : 0;
  const __bf16* p = base + (tile * 16 + (lane & 15)) * 32;
  FragU u;
  u.q[0] = *(const uint4*)(p + kk);        // K kk..kk+7
  u.q[1] = *(const uint4*)(p + kk + 16);   // K kk+16..kk+23
  return u.v;
}

__device__ __forceinline__ v8f wmma_bf16(v16bf a, v16bf b, v8f c) {
  return __builtin_amdgcn_wmma_f32_16x16x32_bf16(false, a, false, b,
                                                 (short)0, c, false, false);
}

// CDNA5 async global -> LDS copy (ASYNCcnt-tracked, no VGPR data path).
__device__ __forceinline__ void async_ld_b128(const void* lds_ptr, const void* gptr) {
  unsigned lds = (unsigned)(uintptr_t)lds_ptr;
  unsigned long long ga = (unsigned long long)(uintptr_t)gptr;
  asm volatile("global_load_async_to_lds_b128 %0, %1, off"
               :: "v"(lds), "v"(ga) : "memory");
}
__device__ __forceinline__ void wait_async0() {
  asm volatile("s_wait_asynccnt 0x0" ::: "memory");
}

// ---------------- kernel 1: zero the output accumulator ----------------
__global__ void zero_out_kernel(float* __restrict__ out, int n) {
  int i = blockIdx.x * blockDim.x + threadIdx.x;
  if (i < n) out[i] = 0.0f;
}

// ---------------- kernel 2: router (one wave32 per token) ----------------
__global__ __launch_bounds__(256)
void router_kernel(const float* __restrict__ x, const float* __restrict__ rw,
                   float* __restrict__ logits, int* __restrict__ topi,
                   float* __restrict__ topw) {
  const int lane  = threadIdx.x & 31;
  const int token = blockIdx.x * 8 + (threadIdx.x >> 5);
  float acc[E_NUM];
#pragma unroll
  for (int e = 0; e < E_NUM; ++e) acc[e] = 0.0f;
  const float* xr = x + (size_t)token * H_DIM;
  for (int h = lane; h < H_DIM; h += 32) {
    float xv = xr[h];
    const float* r = rw + (size_t)h * E_NUM;
#pragma unroll
    for (int e = 0; e < E_NUM; ++e) acc[e] += xv * r[e];
  }
#pragma unroll
  for (int off = 16; off > 0; off >>= 1) {
#pragma unroll
    for (int e = 0; e < E_NUM; ++e) acc[e] += __shfl_xor(acc[e], off, 32);
  }
  if (lane == 0) {
#pragma unroll
    for (int e = 0; e < E_NUM; ++e) logits[(size_t)token * E_NUM + e] = acc[e];
    float m = acc[0];
#pragma unroll
    for (int e = 1; e < E_NUM; ++e) m = fmaxf(m, acc[e]);
    float ex[E_NUM];
#pragma unroll
    for (int e = 0; e < E_NUM; ++e) ex[e] = __expf(acc[e] - m);
    int i0 = 0; float p0 = ex[0];
#pragma unroll
    for (int e = 1; e < E_NUM; ++e) if (ex[e] > p0) { p0 = ex[e]; i0 = e; }
    int i1 = -1; float p1 = -1.0f;
#pragma unroll
    for (int e = 0; e < E_NUM; ++e)
      if (e != i0 && ex[e] > p1) { p1 = ex[e]; i1 = e; }
    float inv = 1.0f / (p0 + p1);
    topi[token * 2 + 0] = i0;  topw[token * 2 + 0] = p0 * inv;
    topi[token * 2 + 1] = i1;  topw[token * 2 + 1] = p1 * inv;
  }
}

// ---------------- kernel 3: deterministic per-expert bucketing ----------------
__global__ __launch_bounds__(1024)
void bucket_kernel(const int* __restrict__ topi, const float* __restrict__ topw,
                   int* __restrict__ counts, int* __restrict__ list_pack,
                   float* __restrict__ list_gate) {
  __shared__ int scan[1024];
  const int tid = threadIdx.x;
  const int t0  = tid * 8;
  int cnt[E_NUM];
#pragma unroll
  for (int e = 0; e < E_NUM; ++e) cnt[e] = 0;
  for (int j = 0; j < 8; ++j) {
    cnt[topi[(t0 + j) * 2 + 0]]++;
    cnt[topi[(t0 + j) * 2 + 1]]++;
  }
  int base[E_NUM];
  for (int e = 0; e < E_NUM; ++e) {
    int val = cnt[e];
    scan[tid] = val;
    __syncthreads();
    for (int off = 1; off < 1024; off <<= 1) {
      int add = (tid >= off) ? scan[tid - off] : 0;
      __syncthreads();
      val += add;
      scan[tid] = val;
      __syncthreads();
    }
    base[e] = val - cnt[e];
    if (tid == 1023) counts[e] = val;
    __syncthreads();
  }
  for (int j = 0; j < 8; ++j) {
    for (int k = 0; k < 2; ++k) {
      int e = topi[(t0 + j) * 2 + k];
      int pos = base[e]++;
      list_pack[e * T_TOK + pos] = (t0 + j) * 2 + k;
      list_gate[e * T_TOK + pos] = topw[(t0 + j) * 2 + k];
    }
  }
}

// ---------------- kernel 4: x -> bf16 ----------------
__global__ void cvt_x_kernel(const float* __restrict__ x, __bf16* __restrict__ xb) {
  int i = blockIdx.x * blockDim.x + threadIdx.x;   // handles 4 elems
  float4 f = ((const float4*)x)[i];
  union { __bf16 h[4]; uint2 u; } p;
  p.h[0] = (__bf16)f.x; p.h[1] = (__bf16)f.y;
  p.h[2] = (__bf16)f.z; p.h[3] = (__bf16)f.w;
  ((uint2*)xb)[i] = p.u;
}

// ---------------- kernel 5: weight transpose + fp32->bf16 ----------------
// W: [K][N] fp32 per expert -> Wt: [N][K] bf16 per expert (k-contiguous rows,
// so GEMM B tiles become contiguous 64B runs suitable for async b128 copies).
__global__ __launch_bounds__(256)
void transpose_cvt_kernel(const float* __restrict__ W, __bf16* __restrict__ Wt,
                          int K, int N) {
  __shared__ __bf16 tile[32][33];   // [k][n], padded
  const float* We = W + (size_t)blockIdx.z * K * N;
  __bf16*     Wte = Wt + (size_t)blockIdx.z * K * N;
  const int n0 = blockIdx.x * 32, k0 = blockIdx.y * 32;
  const int tx = threadIdx.x & 31;
  const int ty = threadIdx.x >> 5;              // 8 rows per pass
  for (int r = ty; r < 32; r += 8)
    tile[r][tx] = (__bf16)We[(size_t)(k0 + r) * N + n0 + tx];
  __syncthreads();
  const int kp = threadIdx.x & 15;              // k pair 0..15
  const int nb = threadIdx.x >> 4;              // 0..15
  for (int h = 0; h < 32; h += 16) {
    int nn = h + nb;
    union { __bf16 b[2]; unsigned u; } pk;
    pk.b[0] = tile[2 * kp][nn];
    pk.b[1] = tile[2 * kp + 1][nn];
    *(unsigned*)&Wte[(size_t)(n0 + nn) * K + k0 + 2 * kp] = pk.u;
  }
}

// ---------------- kernel 6: grouped GEMM1: h = silu(x@w1)*(x@w3)*gate --------
// block = (256-col chunk of I, 64-token tile, expert); 512 threads = 16 waves.
// Double-buffered async global->LDS staging overlapping the WMMA stream.
__global__ __launch_bounds__(512)
void gemm1_kernel(const __bf16* __restrict__ xb,
                  const __bf16* __restrict__ w1t, const __bf16* __restrict__ w3t,
                  const int* __restrict__ counts,
                  const int* __restrict__ list_pack,
                  const float* __restrict__ list_gate,
                  __bf16* __restrict__ hb) {
  const int e  = blockIdx.z;
  const int mb = blockIdx.y;
  const int n0 = blockIdx.x * 256;
  const int cnt = counts[e];
  if (mb * 64 >= cnt) return;
  int mcount = cnt - mb * 64; if (mcount > 64) mcount = 64;

  __shared__ __align__(16) __bf16 As[2][64 * 32];
  __shared__ __align__(16) __bf16 B1[2][256 * 32];
  __shared__ __align__(16) __bf16 B3[2][256 * 32];
  __shared__ int   hrow[64];
  __shared__ int   toksh[64];
  __shared__ float gsh[64];

  const int tid = threadIdx.x;
  if (tid < 64) {
    int pos = mb * 64 + tid;
    if (pos < cnt) {
      int pk = list_pack[e * T_TOK + pos];
      hrow[tid] = pk; toksh[tid] = pk >> 1;
      gsh[tid] = list_gate[e * T_TOK + pos];
    } else { hrow[tid] = 0; toksh[tid] = 0; gsh[tid] = 0.0f; }
  }
  __syncthreads();

  const __bf16* w1te = w1t + (size_t)e * H_DIM * I_DIM;  // [I][H] bf16
  const __bf16* w3te = w3t + (size_t)e * H_DIM * I_DIM;

  // per-thread loop-invariant staging addresses
  const int rr = tid >> 2, cc = (tid & 3) * 8;           // row / 16B-chunk
  const int ldsAoff = rr * 32 + cc;                      // valid when tid<256
  const __bf16* a_src =
      xb + (size_t)toksh[(tid >> 2) & 63] * H_DIM + cc;  // row gather base
  const int ldsBoff0 = rr * 32 + cc;
  const int ldsBoff1 = (rr + 128) * 32 + cc;
  const __bf16* b1p0 = w1te + (size_t)(n0 + rr) * H_DIM + cc;
  const __bf16* b1p1 = b1p0 + (size_t)128 * H_DIM;
  const __bf16* b3p0 = w3te + (size_t)(n0 + rr) * H_DIM + cc;
  const __bf16* b3p1 = b3p0 + (size_t)128 * H_DIM;

  const int lane = tid & 31, w = tid >> 5;
  const int mt = w >> 2, ntb = (w & 3) * 4;

  v8f acc1[4] = {}; v8f acc3[4] = {};

  auto stage = [&](int buf, int k0) {
    if (tid < 256) async_ld_b128(&As[buf][ldsAoff], a_src + k0);
    async_ld_b128(&B1[buf][ldsBoff0], b1p0 + k0);
    async_ld_b128(&B1[buf][ldsBoff1], b1p1 + k0);
    async_ld_b128(&B3[buf][ldsBoff0], b3p0 + k0);
    async_ld_b128(&B3[buf][ldsBoff1], b3p1 + k0);
  };

  stage(0, 0);
  const int NSTEP = H_DIM / 32;
  for (int ks = 0; ks < NSTEP; ++ks) {
    wait_async0();        // this wave's stage ks complete (only one in flight)
    __syncthreads();      // all waves' stage ks done; all reads of other buf done
    if (ks + 1 < NSTEP) stage((ks + 1) & 1, (ks + 1) * 32);
    const __bf16* Ab  = As[ks & 1];
    const __bf16* B1b = B1[ks & 1];
    const __bf16* B3b = B3[ks & 1];
    v16bf a = ld_frag16(Ab, mt);
#pragma unroll
    for (int j = 0; j < 4; ++j) {
      v16bf b1 = ld_frag16(B1b, ntb + j);
      acc1[j] = wmma_bf16(a, b1, acc1[j]);
      v16bf b3 = ld_frag16(B3b, ntb + j);
      acc3[j] = wmma_bf16(a, b3, acc3[j]);
    }
  }

  // epilogue: h = silu(a1)*a3*gate (gate folded here; GEMM2 is ungated)
  const int roff = (lane & 16) ? 8 : 0;
#pragma unroll
  for (int j = 0; j < 4; ++j) {
#pragma unroll
    for (int v = 0; v < 8; ++v) {
      int rl = mt * 16 + v + roff;
      if (rl < mcount) {
        float c1 = acc1[j][v];
        float sig = __builtin_amdgcn_rcpf(1.0f + __expf(-c1));
        float h = c1 * sig * acc3[j][v] * gsh[rl];
        int col = n0 + (ntb + j) * 16 + (lane & 15);
        hb[(size_t)hrow[rl] * I_DIM + col] = (__bf16)h;
      }
    }
  }
}

// ---------------- kernel 7: grouped GEMM2: out += h @ w2[e] ----------------
__global__ __launch_bounds__(512)
void gemm2_kernel(const __bf16* __restrict__ hb, const __bf16* __restrict__ w2t,
                  const int* __restrict__ counts,
                  const int* __restrict__ list_pack,
                  float* __restrict__ out) {
  const int e  = blockIdx.z;
  const int mb = blockIdx.y;
  const int n0 = blockIdx.x * 256;
  const int cnt = counts[e];
  if (mb * 64 >= cnt) return;
  int mcount = cnt - mb * 64; if (mcount > 64) mcount = 64;

  __shared__ __align__(16) __bf16 As[2][64 * 32];
  __shared__ __align__(16) __bf16 Bt[2][256 * 32];
  __shared__ int hrow[64];
  __shared__ int toksh[64];

  const int tid = threadIdx.x;
  if (tid < 64) {
    int pos = mb * 64 + tid;
    if (pos < cnt) {
      int pk = list_pack[e * T_TOK + pos];
      hrow[tid] = pk; toksh[tid] = pk >> 1;
    } else { hrow[tid] = 0; toksh[tid] = 0; }
  }
  __syncthreads();

  const __bf16* w2te = w2t + (size_t)e * I_DIM * H_DIM;  // [H][I] bf16

  const int rr = tid >> 2, cc = (tid & 3) * 8;
  const int ldsAoff = rr * 32 + cc;
  const __bf16* a_src =
      hb + (size_t)hrow[(tid >> 2) & 63] * I_DIM + cc;
  const int ldsBoff0 = rr * 32 + cc;
  const int ldsBoff1 = (rr + 128) * 32 + cc;
  const __bf16* bp0 = w2te + (size_t)(n0 + rr) * I_DIM + cc;
  const __bf16* bp1 = bp0 + (size_t)128 * I_DIM;

  const int lane = tid & 31, w = tid >> 5;
  const int mt = w >> 2, ntb = (w & 3) * 4;

  v8f acc[4] = {};

  auto stage = [&](int buf, int k0) {
    if (tid < 256) async_ld_b128(&As[buf][ldsAoff], a_src + k0);
    async_ld_b128(&Bt[buf][ldsBoff0], bp0 + k0);
    async_ld_b128(&Bt[buf][ldsBoff1], bp1 + k0);
  };

  stage(0, 0);
  const int NSTEP = I_DIM / 32;
  for (int ks = 0; ks < NSTEP; ++ks) {
    wait_async0();
    __syncthreads();
    if (ks + 1 < NSTEP) stage((ks + 1) & 1, (ks + 1) * 32);
    const __bf16* Ab = As[ks & 1];
    const __bf16* Bb = Bt[ks & 1];
    v16bf a = ld_frag16(Ab, mt);
#pragma unroll
    for (int j = 0; j < 4; ++j) {
      v16bf b = ld_frag16(Bb, ntb + j);
      acc[j] = wmma_bf16(a, b, acc[j]);
    }
  }

  const int roff = (lane & 16) ? 8 : 0;
#pragma unroll
  for (int j = 0; j < 4; ++j) {
#pragma unroll
    for (int v = 0; v < 8; ++v) {
      int rl = mt * 16 + v + roff;
      if (rl < mcount) {
        int col = n0 + (ntb + j) * 16 + (lane & 15);
        atomicAdd(&out[(size_t)toksh[rl] * H_DIM + col], acc[j][v]);
      }
    }
  }
}

// ---------------- host launcher ----------------
extern "C" void kernel_launch(void* const* d_in, const int* in_sizes, int n_in,
                              void* d_out, int out_size, void* d_ws, size_t ws_size,
                              hipStream_t stream) {
  const float* x  = (const float*)d_in[0];   // [B,S,H]
  const float* rw = (const float*)d_in[1];   // [H,E]
  const float* w1 = (const float*)d_in[2];   // [E,H,I]
  const float* w3 = (const float*)d_in[3];   // [E,H,I]
  const float* w2 = (const float*)d_in[4];   // [E,I,H]

  float* out    = (float*)d_out;                        // [T,H]
  float* logits = out + (size_t)T_TOK * H_DIM;          // [T,E]

  // workspace layout (~177 MB total)
  char* ws = (char*)d_ws;
  __bf16* hb  = (__bf16*)ws;             ws += (size_t)2 * T_TOK * I_DIM * 2; // 64MB
  __bf16* xb  = (__bf16*)ws;             ws += (size_t)T_TOK * H_DIM * 2;     // 16MB
  __bf16* w1t = (__bf16*)ws;             ws += (size_t)E_NUM * H_DIM * I_DIM * 2; // 32MB
  __bf16* w3t = (__bf16*)ws;             ws += (size_t)E_NUM * H_DIM * I_DIM * 2; // 32MB
  __bf16* w2t = (__bf16*)ws;             ws += (size_t)E_NUM * H_DIM * I_DIM * 2; // 32MB
  int*   topi      = (int*)ws;           ws += (size_t)T_TOK * 2 * 4;
  float* topw      = (float*)ws;         ws += (size_t)T_TOK * 2 * 4;
  int*   counts    = (int*)ws;           ws += 256;
  int*   list_pack = (int*)ws;           ws += (size_t)E_NUM * T_TOK * 4;
  float* list_gate = (float*)ws;         /* ws += E*T*4 */

  const int nz = T_TOK * H_DIM;
  zero_out_kernel<<<(nz + 255) / 256, 256, 0, stream>>>(out, nz);
  router_kernel<<<T_TOK / 8, 256, 0, stream>>>(x, rw, logits, topi, topw);
  bucket_kernel<<<1, 1024, 0, stream>>>(topi, topw, counts, list_pack, list_gate);
  cvt_x_kernel<<<(T_TOK * H_DIM / 4 + 255) / 256, 256, 0, stream>>>(x, xb);

  // per-launch weight transform: fp32 [K][N] -> bf16 [N][K]
  {
    dim3 gA(I_DIM / 32, H_DIM / 32, E_NUM);   // w1,w3: K=H, N=I
    transpose_cvt_kernel<<<gA, 256, 0, stream>>>(w1, w1t, H_DIM, I_DIM);
    transpose_cvt_kernel<<<gA, 256, 0, stream>>>(w3, w3t, H_DIM, I_DIM);
    dim3 gB(H_DIM / 32, I_DIM / 32, E_NUM);   // w2: K=I, N=H
    transpose_cvt_kernel<<<gB, 256, 0, stream>>>(w2, w2t, I_DIM, H_DIM);
  }

  dim3 g1(I_DIM / 256, T_TOK / 64, E_NUM);
  gemm1_kernel<<<g1, 512, 0, stream>>>(xb, w1t, w3t, counts, list_pack, list_gate, hb);
  dim3 g2(H_DIM / 256, T_TOK / 64, E_NUM);
  gemm2_kernel<<<g2, 512, 0, stream>>>(hb, w2t, counts, list_pack, out);
}
